// EdgeConv_56246891708812
// MI455X (gfx1250) — matrix operand order repaired and checked by reading, not compiled, not run on hardware
//
#include <hip/hip_runtime.h>

// EdgeConv for MI455X (gfx1250): bf16 WMMA 16x16x32, 3-pass recompute scheme.
// B=2, C=64, N=16384, K=32, H1=H2=64.

typedef __attribute__((ext_vector_type(16))) __bf16 v16bf;
typedef __attribute__((ext_vector_type(2)))  __bf16 v2bf;
typedef __attribute__((ext_vector_type(8)))  float  v8f;

#define B_  2
#define C_  64
#define N_  16384
#define K_  32
#define PTOTF 1048576.0f   // B*N*K positions per BN reduction

union BF16x16 {
  v16bf v;
  uint4 u[2];
};

__device__ __forceinline__ unsigned int pack_bf16(float a, float b) {
  v2bf p;
  p.x = (__bf16)a;
  p.y = (__bf16)b;
  return __builtin_bit_cast(unsigned int, p);
}

__device__ __forceinline__ float lrelu(float v) { return v >= 0.0f ? v : 0.2f * v; }

// ds_swizzle xor-butterfly within 16-lane groups (masks 1,2,4,8; and=0x1f keeps
// group-of-32 mode; xor<=8 never crosses a 16-lane half).
#define FSWZ(x, pat) __uint_as_float(__builtin_amdgcn_ds_swizzle(__float_as_uint(x), pat))

__device__ __forceinline__ float bfly_sum16(float v) {
  v += FSWZ(v, 0x041F);
  v += FSWZ(v, 0x081F);
  v += FSWZ(v, 0x101F);
  v += FSWZ(v, 0x201F);
  return v;
}

__device__ __forceinline__ float bfly_max16(float v) {
  v = fmaxf(v, FSWZ(v, 0x041F));
  v = fmaxf(v, FSWZ(v, 0x081F));
  v = fmaxf(v, FSWZ(v, 0x101F));
  v = fmaxf(v, FSWZ(v, 0x201F));
  return v;
}

// LDS layout (dword offsets)
#define W1L_OFF 0        // 64 rows * 68 dwords (bf16-pair packed W1, padded stride)
#define W2L_OFF 4352     // 64 rows * 36 dwords
#define STAT_OFF 6656    // 128 floats (per-block sum/sumsq)
#define EL_OFF  6784     // 8 waves * 2048 dwords: edge tile 128x32 bf16, xor-swizzled
#define HL_OFF  23168    // 8 waves * 1024 dwords: h1 tile 64x32 bf16, xor-swizzled
#define LDS_DWORDS 31360 // 125440 bytes < 320KB/WGP

// ws (float) layout: [0..63]=sum1 [64..127]=sumsq1 [128..191]=sum2 [192..255]=sumsq2
//                    [256..319]=scale1 [320..383]=shift1 [384..447]=scale2 [448..511]=shift2

// MODE 0: layer1 + stats1.  MODE 1: layer1+BN1+layer2 + stats2.  MODE 2: full + max + store.
template<int MODE>
__global__ void __launch_bounds__(256)
edgeconv_kernel(const float* __restrict__ x, const int* __restrict__ knn,
                const float* __restrict__ W1, const float* __restrict__ W2,
                float* __restrict__ ws, float* __restrict__ out,
                int points_per_wave)
{
  extern __shared__ unsigned int lds[];
  unsigned int* W1l = lds + W1L_OFF;
  unsigned int* W2l = lds + W2L_OFF;
  float* statsl = (float*)(lds + STAT_OFF);
  const int tid  = threadIdx.x;
  const int lane = tid & 31;
  const int wv   = tid >> 5;
  const int lh   = lane >> 4;     // lane half (WMMA K sub-block select)
  const int l15  = lane & 15;
  unsigned int* El = lds + EL_OFF + wv * 2048;
  unsigned int* Hl = lds + HL_OFF + wv * 1024;

  if (MODE < 2 && tid < 128) statsl[tid] = 0.0f;

  // Convert weights fp32 -> packed bf16 pairs in LDS (row-major over input channel).
  for (int i = tid; i < 64 * 64; i += 256) {
    int o = i >> 6, j = i & 63;
    W1l[o * 68 + j] = pack_bf16(W1[o * 128 + 2 * j], W1[o * 128 + 2 * j + 1]);
  }
  if (MODE >= 1) {
    for (int i = tid; i < 64 * 32; i += 256) {
      int o = i >> 5, j = i & 31;
      W2l[o * 36 + j] = pack_bf16(W2[o * 64 + 2 * j], W2[o * 64 + 2 * j + 1]);
    }
  }
  __syncthreads();

  // Per-wave deferred BN statistics accumulators (sum / sumsq per channel-tile),
  // reduced across lanes only once at the end of the point loop.
  v8f ssum[4], ssq[4];
  #pragma unroll
  for (int i = 0; i < 4; ++i) { ssum[i] = (v8f){}; ssq[i] = (v8f){}; }

  const int pstart = (blockIdx.x * 8 + wv) * points_per_wave;
  for (int it = 0; it < points_per_wave; ++it) {
    const int p = pstart + it;          // p = b*N + n
    const int b = __builtin_amdgcn_readfirstlane(p >> 14);
    const int n = __builtin_amdgcn_readfirstlane(p & (N_ - 1));
    const float* xb  = x + b * (C_ * N_);
    const float* xcp = xb + n;            // wave-uniform center point -> scalar loads
    const int ik = knn[(p << 5) + lane];  // neighbor index for this lane's k (divergent)

    // ---- Build edge-feature tile E (128 x 32) bf16 in LDS; lane owns column k=lane.
    // Channels 0..63 = x[nbr]-x[ctr]; 64..127 = x[ctr].  16B-group xor swizzle.
    #pragma unroll 4
    for (int g = 0; g < 16; ++g) {
      unsigned int uu[4];
      #pragma unroll
      for (int q = 0; q < 4; ++q) {
        const int j = g * 4 + q;
        float e0, e1;
        if (j < 32) {
          const int c = j * 2;
          e0 = xb[c * N_ + ik] - xcp[c * N_];
          e1 = xb[(c + 1) * N_ + ik] - xcp[(c + 1) * N_];
        } else {
          const int c = j * 2 - 64;
          e0 = xcp[c * N_];
          e1 = xcp[(c + 1) * N_];
        }
        uu[q] = pack_bf16(e0, e1);
      }
      *reinterpret_cast<uint4*>(El + lane * 64 + ((g ^ l15) << 2)) =
          make_uint4(uu[0], uu[1], uu[2], uu[3]);
    }
    asm volatile("s_wait_dscnt 0x0" ::: "memory");

    // ---- Layer 1: 4 channel tiles x 2 k-column tiles, K=128 -> 4 WMMA steps each.
    #pragma unroll
    for (int mt = 0; mt < 4; ++mt) {
      const unsigned int* wrow = W1l + (mt * 16 + l15) * 68;
      BF16x16 afr[4];
      #pragma unroll
      for (int t = 0; t < 4; ++t) {   // A frag: K = 32t + 16*lh + {0..15}
        afr[t].u[0] = *(const uint4*)(wrow + 16 * t + 8 * lh);
        afr[t].u[1] = *(const uint4*)(wrow + 16 * t + 4 + 8 * lh);
      }
      v8f acc[2];
      #pragma unroll
      for (int ct = 0; ct < 2; ++ct) {
        const unsigned int* colp = El + (ct * 16 + l15) * 64;
        v8f c = {};
        #pragma unroll
        for (int t = 0; t < 4; ++t) { // B frag: col=lane%16, K = 32t + 16*lh + {0..15}
          BF16x16 bf;
          const int g0 = 4 * t + 2 * lh;
          bf.u[0] = *(const uint4*)(colp + ((g0 ^ l15) << 2));
          bf.u[1] = *(const uint4*)(colp + (((g0 + 1) ^ l15) << 2));
          c = __builtin_amdgcn_wmma_f32_16x16x32_bf16(false, afr[t].v, false, bf.v,
                                                      (short)0, c, false, false);
        }
        acc[ct] = c;
      }

      if (MODE == 0) {
        // accumulate per-(channel,column) partial sums; cross-lane reduce at end
        ssum[mt] += acc[0] + acc[1];
        ssq[mt]  += acc[0] * acc[0] + acc[1] * acc[1];
      } else {
        // BN1 + LeakyReLU, repack C-frag channel pairs into layer-2 B layout in LDS
        #pragma unroll
        for (int r = 0; r < 8; r += 2) {
          const int ch = mt * 16 + lh * 8 + r;
          const float s0 = ws[256 + ch],     t0 = ws[320 + ch];
          const float s1 = ws[256 + ch + 1], t1 = ws[320 + ch + 1];
          #pragma unroll
          for (int ct = 0; ct < 2; ++ct) {
            const float y0 = lrelu(fmaf(acc[ct][r],     s0, t0));
            const float y1 = lrelu(fmaf(acc[ct][r + 1], s1, t1));
            const int colix = ct * 16 + l15;
            Hl[colix * 32 + ((((mt * 2 + lh) ^ (colix & 7)) << 2) | (r >> 1))] = pack_bf16(y0, y1);
          }
        }
      }
    }

    // ---- Layer 2 (K=64 -> 2 WMMA steps), stats2 or BN2 + max-over-k + store.
    if (MODE >= 1) {
      asm volatile("s_wait_dscnt 0x0" ::: "memory");
      #pragma unroll
      for (int mt = 0; mt < 4; ++mt) {
        const unsigned int* wrow = W2l + (mt * 16 + l15) * 36;
        BF16x16 afr[2];
        #pragma unroll
        for (int t = 0; t < 2; ++t) {
          afr[t].u[0] = *(const uint4*)(wrow + 16 * t + 8 * lh);
          afr[t].u[1] = *(const uint4*)(wrow + 16 * t + 4 + 8 * lh);
        }
        v8f acc[2];
        #pragma unroll
        for (int ct = 0; ct < 2; ++ct) {
          const unsigned int* colp = Hl + (ct * 16 + l15) * 32;
          v8f c = {};
          #pragma unroll
          for (int t = 0; t < 2; ++t) {
            BF16x16 bf;
            const int g0 = 4 * t + 2 * lh;
            const int key = l15 & 7;
            bf.u[0] = *(const uint4*)(colp + ((g0 ^ key) << 2));
            bf.u[1] = *(const uint4*)(colp + (((g0 + 1) ^ key) << 2));
            c = __builtin_amdgcn_wmma_f32_16x16x32_bf16(false, afr[t].v, false, bf.v,
                                                        (short)0, c, false, false);
          }
          acc[ct] = c;
        }

        if (MODE == 1) {
          ssum[mt] += acc[0] + acc[1];
          ssq[mt]  += acc[0] * acc[0] + acc[1] * acc[1];
        } else {
          #pragma unroll
          for (int r = 0; r < 8; ++r) {
            const int ch = mt * 16 + lh * 8 + r;
            const float sc = ws[384 + ch], sh = ws[448 + ch];
            float v = fmaxf(lrelu(fmaf(acc[0][r], sc, sh)),
                            lrelu(fmaf(acc[1][r], sc, sh)));
            v = bfly_max16(v);
            if (l15 == r) out[(b * 64 + ch) * N_ + n] = v;
          }
        }
      }
      asm volatile("s_wait_dscnt 0x0" ::: "memory");
    }
  }

  if (MODE < 2) {
    // one cross-lane reduction per wave, then block-level flush to global
    #pragma unroll
    for (int mt = 0; mt < 4; ++mt) {
      #pragma unroll
      for (int r = 0; r < 8; ++r) {
        const float s  = bfly_sum16(ssum[mt][r]);
        const float sq = bfly_sum16(ssq[mt][r]);
        if (l15 == r) {
          const int ch = mt * 16 + lh * 8 + r;
          atomicAdd(&statsl[ch], s);
          atomicAdd(&statsl[64 + ch], sq);
        }
      }
    }
    __syncthreads();
    if (tid < 128) atomicAdd(&ws[(MODE == 0 ? 0 : 128) + tid], statsl[tid]);
  }
}

__global__ void zero_stats_kernel(float* ws) { ws[threadIdx.x] = 0.0f; }  // 256 threads

__global__ void finalize_kernel(const float* __restrict__ gamma,
                                const float* __restrict__ beta,
                                float* __restrict__ ws, int which) {
  const int c = threadIdx.x;  // 64 threads
  const float sum = ws[which * 128 + c];
  const float sq  = ws[which * 128 + 64 + c];
  const float mu  = sum * (1.0f / PTOTF);
  const float var = sq * (1.0f / PTOTF) - mu * mu;
  const float sc  = gamma[c] * rsqrtf(var + 1e-5f);
  ws[256 + which * 128 + c]      = sc;
  ws[256 + which * 128 + 64 + c] = beta[c] - mu * sc;
}

extern "C" void kernel_launch(void* const* d_in, const int* in_sizes, int n_in,
                              void* d_out, int out_size, void* d_ws, size_t ws_size,
                              hipStream_t stream) {
  (void)in_sizes; (void)n_in; (void)out_size; (void)ws_size;
  const float* x  = (const float*)d_in[0];
  const int*  knn = (const int*)d_in[1];
  const float* W1 = (const float*)d_in[2];
  const float* g1 = (const float*)d_in[3];
  const float* b1 = (const float*)d_in[4];
  const float* W2 = (const float*)d_in[5];
  const float* g2 = (const float*)d_in[6];
  const float* b2 = (const float*)d_in[7];
  float* ws  = (float*)d_ws;
  float* out = (float*)d_out;

  const size_t shmem = LDS_DWORDS * sizeof(unsigned int);  // 125440 B
  const int ppw = 8;
  const int blocks = (B_ * N_) / (8 * ppw);  // 512 blocks * 8 waves * 8 points = 32768

  zero_stats_kernel<<<1, 256, 0, stream>>>(ws);
  edgeconv_kernel<0><<<blocks, 256, shmem, stream>>>(x, knn, W1, W2, ws, out, ppw);
  finalize_kernel<<<1, 64, 0, stream>>>(g1, b1, ws, 0);
  edgeconv_kernel<1><<<blocks, 256, shmem, stream>>>(x, knn, W1, W2, ws, out, ppw);
  finalize_kernel<<<1, 64, 0, stream>>>(g2, b2, ws, 1);
  edgeconv_kernel<2><<<blocks, 256, shmem, stream>>>(x, knn, W1, W2, ws, out, ppw);
}
